// LinearAttention_24541443129828
// MI455X (gfx1250) — matrix operand order repaired and checked by reading, not compiled, not run on hardware
//
#include <hip/hip_runtime.h>
#include <hip/hip_bf16.h>

// Linear attention (causal), chunk-parallel scan formulation for MI455X.
// B=1, H=16, N=2048, D=Dv=64. fp32 in/out, f16 WMMA compute w/ f32 accum.
// All WMMA operands staged in LDS so both A and B fragment loads are
// contiguous 16-byte ds_load_b128 accesses.

#define N_SEQ 2048
#define DHEAD 64
#define CHUNK 64
#define NCHUNK (N_SEQ / CHUNK)
#define HEADS 16
#define EPSV 1e-6f
#define LDT 72   // padded leading dim (halfs) for transpose-stored arrays

typedef __attribute__((ext_vector_type(16))) _Float16 v16h;
typedef __attribute__((ext_vector_type(8)))  _Float16 v8h;
typedef __attribute__((ext_vector_type(8)))  float    v8f;

__device__ __forceinline__ float phi(float x) {
    // elu(x) + 1
    return x > 0.0f ? x + 1.0f : __expf(x);
}

// ---- CDNA5 wave32 WMMA fragment loaders (CDNA5 ISA layouts) ----
// All loads are 16-byte contiguous -> ds_load_b128.

// A-matrix 16x32 f16 (MxK) from M-major storage src[m*ld + k].
// lane L: M=L%16; element e=2v+h: K = 16*(v>>2) + 8*(L>=16) + 2*(v&3)+h
// => two contiguous 8-half runs at k = khi and k = 16+khi.
__device__ __forceinline__ v16h load_A_frag(const _Float16* base, int ld) {
    const int lane = threadIdx.x & 31;
    const int m    = lane & 15;
    const int khi  = (lane >> 4) << 3; // 0 or 8
    const _Float16* p = base + m * ld;
    v8h lo = *(const v8h*)(p + khi);
    v8h hi = *(const v8h*)(p + 16 + khi);
    return __builtin_shufflevector(lo, hi, 0, 1, 2, 3, 4, 5, 6, 7,
                                   8, 9, 10, 11, 12, 13, 14, 15);
}

// B-matrix 32x16 f16 (KxN) from N-major storage src[n*ld + k].
// lane L: N=L%16; element e: K = 16*(e>>3) + 8*(L>=16) + (e&7)
// => two contiguous 8-half runs at k = koff and k = 16+koff.
__device__ __forceinline__ v16h load_B_frag(const _Float16* base, int ld) {
    const int lane = threadIdx.x & 31;
    const int n    = lane & 15;
    const int koff = (lane >> 4) << 3; // 0 or 8
    const _Float16* p = base + n * ld + koff;
    v8h lo = *(const v8h*)(p);
    v8h hi = *(const v8h*)(p + 16);
    return __builtin_shufflevector(lo, hi, 0, 1, 2, 3, 4, 5, 6, 7,
                                   8, 9, 10, 11, 12, 13, 14, 15);
}

// ---------------------------------------------------------------------------
// Kernel 1: per-chunk sums  Sc[h][c] = phi(K_c)^T V_c  (64x64),
//           zc[h][c][d]    = sum_m phi(K_c)[m][d]
// grid (NCHUNK, HEADS), 128 threads (4 waves). Wave w owns d-rows [16w,16w+16).
// ---------------------------------------------------------------------------
__global__ void linattn_k1_chunk_stats(const float* __restrict__ K,
                                       const float* __restrict__ V,
                                       float* __restrict__ Sc,
                                       float* __restrict__ zc) {
    const int c = blockIdx.x, h = blockIdx.y;
    __shared__ __align__(16) _Float16 kfT[DHEAD * LDT]; // [d][m] = phi(K)[m][d]  (A: M=d,K=m)
    __shared__ __align__(16) _Float16 vsT[DHEAD * LDT]; // [e][m] = V[m][e]       (B: K=m,N=e)

    const int tid = threadIdx.x;
    const size_t base = ((size_t)h * N_SEQ + (size_t)c * CHUNK) * DHEAD;
    const float4* K4 = (const float4*)(K + base);
    const float4* V4 = (const float4*)(V + base);

    for (int i = tid; i < (CHUNK * DHEAD) / 4; i += blockDim.x) {
        const int m = i >> 4;            // element row
        const int d = (i & 15) << 2;     // element col (4 consecutive)
        const float4 kv = K4[i];
        const float4 vv = V4[i];
        kfT[(d + 0) * LDT + m] = (_Float16)phi(kv.x);
        kfT[(d + 1) * LDT + m] = (_Float16)phi(kv.y);
        kfT[(d + 2) * LDT + m] = (_Float16)phi(kv.z);
        kfT[(d + 3) * LDT + m] = (_Float16)phi(kv.w);
        vsT[(d + 0) * LDT + m] = (_Float16)vv.x;
        vsT[(d + 1) * LDT + m] = (_Float16)vv.y;
        vsT[(d + 2) * LDT + m] = (_Float16)vv.z;
        vsT[(d + 3) * LDT + m] = (_Float16)vv.w;
    }
    __syncthreads();

    const int wave = tid >> 5, lane = tid & 31;
    const int n = lane & 15, grp = lane >> 4;

    // Hoisted A fragments: rows d in [16w,16w+16), K = m in [0,32) and [32,64)
    const v16h a0 = load_A_frag(&kfT[(16 * wave) * LDT + 0],  LDT);
    const v16h a1 = load_A_frag(&kfT[(16 * wave) * LDT + 32], LDT);

    float* Sout = Sc + ((size_t)(h * NCHUNK + c)) * DHEAD * DHEAD;
#pragma unroll
    for (int j = 0; j < 4; ++j) {
        v8f acc = {};
        acc = __builtin_amdgcn_wmma_f32_16x16x32_f16(
            false, a0, false, load_B_frag(&vsT[(16 * j) * LDT + 0], LDT),
            (short)0, acc, false, false);
        acc = __builtin_amdgcn_wmma_f32_16x16x32_f16(
            false, a1, false, load_B_frag(&vsT[(16 * j) * LDT + 32], LDT),
            (short)0, acc, false, false);
#pragma unroll
        for (int r = 0; r < 8; ++r) {
            const int drow = 16 * wave + r + 8 * grp; // M = r + 8*(lane>=16)
            Sout[(size_t)drow * DHEAD + 16 * j + n] = acc[r];
        }
    }

    if (tid < DHEAD) {
        float s = 0.0f;
        for (int m = 0; m < CHUNK; ++m) s += (float)kfT[tid * LDT + m];
        zc[(size_t)(h * NCHUNK + c) * DHEAD + tid] = s;
    }
}

// ---------------------------------------------------------------------------
// Kernel 2: exclusive prefix scan over chunks (in place) for Sc and zc.
// grid (HEADS), 256 threads; element-parallel, serial over 32 chunks.
// ---------------------------------------------------------------------------
__global__ void linattn_k2_scan(float* __restrict__ Sc, float* __restrict__ zc) {
    const int h = blockIdx.x;
    const int SE = DHEAD * DHEAD;
    for (int e = threadIdx.x; e < SE + DHEAD; e += blockDim.x) {
        float run = 0.0f;
        if (e < SE) {
            for (int c = 0; c < NCHUNK; ++c) {
                float* p = Sc + ((size_t)(h * NCHUNK + c)) * SE + e;
                const float t = *p; *p = run; run += t;
            }
        } else {
            const int d = e - SE;
            for (int c = 0; c < NCHUNK; ++c) {
                float* p = zc + (size_t)(h * NCHUNK + c) * DHEAD + d;
                const float t = *p; *p = run; run += t;
            }
        }
    }
}

// ---------------------------------------------------------------------------
// Kernel 3: per-chunk output.
//   A      = mask(phi(Q_c) phi(K_c)^T)            (WMMA, 64x64)
//   den[m] = phi(Q_c)[m] . z_prefix + rowsum(A[m]) + eps
//   O      = (phi(Q_c) @ S_prefix + A @ V_c) / den
// grid (NCHUNK, HEADS), 128 threads (4 waves). Wave w owns rows [16w,16w+16).
// ---------------------------------------------------------------------------
__global__ void linattn_k3_output(const float* __restrict__ Q,
                                  const float* __restrict__ K,
                                  const float* __restrict__ V,
                                  const float* __restrict__ Sc,
                                  const float* __restrict__ zc,
                                  float* __restrict__ Out) {
    const int c = blockIdx.x, h = blockIdx.y;
    __shared__ __align__(16) _Float16 qf[CHUNK * DHEAD];  // [m][d] phi(Q)        (A: M=m,K=d)
    __shared__ __align__(16) _Float16 kf[CHUNK * DHEAD];  // [m][d] phi(K)        (B: K=d,N=m)
    __shared__ __align__(16) _Float16 vsT[DHEAD * LDT];   // [e][m] V             (B: K=m,N=e)
    __shared__ __align__(16) _Float16 spT[DHEAD * LDT];   // [e][d] prefix state  (B: K=d,N=e)
    __shared__ __align__(16) _Float16 am[CHUNK * DHEAD];  // [m][mc] masked A     (A: M=m,K=mc)
    __shared__ float zp[DHEAD];
    __shared__ float den[CHUNK];

    const int tid = threadIdx.x;
    const size_t base = ((size_t)h * N_SEQ + (size_t)c * CHUNK) * DHEAD;
    const float4* Q4 = (const float4*)(Q + base);
    const float4* K4 = (const float4*)(K + base);
    const float4* V4 = (const float4*)(V + base);
    const float4* S4 = (const float4*)(Sc + ((size_t)(h * NCHUNK + c)) * DHEAD * DHEAD);

    for (int i = tid; i < (CHUNK * DHEAD) / 4; i += blockDim.x) {
        const int m = i >> 4;
        const int d = (i & 15) << 2;
        const float4 qv = Q4[i];
        const float4 kv = K4[i];
        const float4 vv = V4[i];
        const float4 sv = S4[i];
        // row-major stores (contiguous -> ds_store_b64)
        qf[m * DHEAD + d + 0] = (_Float16)phi(qv.x);
        qf[m * DHEAD + d + 1] = (_Float16)phi(qv.y);
        qf[m * DHEAD + d + 2] = (_Float16)phi(qv.z);
        qf[m * DHEAD + d + 3] = (_Float16)phi(qv.w);
        kf[m * DHEAD + d + 0] = (_Float16)phi(kv.x);
        kf[m * DHEAD + d + 1] = (_Float16)phi(kv.y);
        kf[m * DHEAD + d + 2] = (_Float16)phi(kv.z);
        kf[m * DHEAD + d + 3] = (_Float16)phi(kv.w);
        // transposed stores (padded ld)
        vsT[(d + 0) * LDT + m] = (_Float16)vv.x;
        vsT[(d + 1) * LDT + m] = (_Float16)vv.y;
        vsT[(d + 2) * LDT + m] = (_Float16)vv.z;
        vsT[(d + 3) * LDT + m] = (_Float16)vv.w;
        spT[(d + 0) * LDT + m] = (_Float16)sv.x;  // here m indexes state row d', d indexes e
        spT[(d + 1) * LDT + m] = (_Float16)sv.y;
        spT[(d + 2) * LDT + m] = (_Float16)sv.z;
        spT[(d + 3) * LDT + m] = (_Float16)sv.w;
    }
    if (tid < DHEAD) zp[tid] = zc[(size_t)(h * NCHUNK + c) * DHEAD + tid];
    __syncthreads();

    const int wave = tid >> 5, lane = tid & 31;
    const int n = lane & 15, grp = lane >> 4;

    // Hoisted Qf A fragments for this wave's row strip.
    const v16h aq0 = load_A_frag(&qf[(16 * wave) * DHEAD + 0],  DHEAD);
    const v16h aq1 = load_A_frag(&qf[(16 * wave) * DHEAD + 32], DHEAD);

    // Phase 1: intra-chunk scores, causal mask, stash to LDS as f16.
#pragma unroll
    for (int j = 0; j < 4; ++j) {
        v8f acc = {};
        acc = __builtin_amdgcn_wmma_f32_16x16x32_f16(
            false, aq0, false, load_B_frag(&kf[(16 * j) * DHEAD + 0], DHEAD),
            (short)0, acc, false, false);
        acc = __builtin_amdgcn_wmma_f32_16x16x32_f16(
            false, aq1, false, load_B_frag(&kf[(16 * j) * DHEAD + 32], DHEAD),
            (short)0, acc, false, false);
#pragma unroll
        for (int r = 0; r < 8; ++r) {
            const int row = 16 * wave + r + 8 * grp;
            const int col = 16 * j + n;
            const float v = (col <= row) ? acc[r] : 0.0f;
            am[row * DHEAD + col] = (_Float16)v;
        }
    }
    __syncthreads();

    // Denominator per row.
    if (tid < CHUNK) {
        float s = 0.0f;
        for (int d = 0; d < DHEAD; ++d) s += (float)qf[tid * DHEAD + d] * zp[d];
        for (int mm = 0; mm <= tid; ++mm) s += (float)am[tid * DHEAD + mm];
        den[tid] = s + EPSV;
    }
    __syncthreads();

    // Hoisted masked-A fragments for this wave's row strip.
    const v16h aa0 = load_A_frag(&am[(16 * wave) * DHEAD + 0],  DHEAD);
    const v16h aa1 = load_A_frag(&am[(16 * wave) * DHEAD + 32], DHEAD);

    // Phase 2: O = qf @ sp + am @ vs, then divide by den.
    float* Oc = Out + base; // Dv == D
#pragma unroll
    for (int j = 0; j < 4; ++j) {
        v8f acc = {};
        acc = __builtin_amdgcn_wmma_f32_16x16x32_f16(
            false, aq0, false, load_B_frag(&spT[(16 * j) * LDT + 0], LDT),
            (short)0, acc, false, false);
        acc = __builtin_amdgcn_wmma_f32_16x16x32_f16(
            false, aq1, false, load_B_frag(&spT[(16 * j) * LDT + 32], LDT),
            (short)0, acc, false, false);
        acc = __builtin_amdgcn_wmma_f32_16x16x32_f16(
            false, aa0, false, load_B_frag(&vsT[(16 * j) * LDT + 0], LDT),
            (short)0, acc, false, false);
        acc = __builtin_amdgcn_wmma_f32_16x16x32_f16(
            false, aa1, false, load_B_frag(&vsT[(16 * j) * LDT + 32], LDT),
            (short)0, acc, false, false);
#pragma unroll
        for (int r = 0; r < 8; ++r) {
            const int row = 16 * wave + r + 8 * grp;
            Oc[(size_t)row * DHEAD + 16 * j + n] = acc[r] / den[row];
        }
    }
}

extern "C" void kernel_launch(void* const* d_in, const int* in_sizes, int n_in,
                              void* d_out, int out_size, void* d_ws, size_t ws_size,
                              hipStream_t stream) {
    const float* Q = (const float*)d_in[0];
    const float* K = (const float*)d_in[1];
    const float* V = (const float*)d_in[2];
    // d_in[3] = causal flag (setup uses causal=1; causal path implemented)
    float* Out = (float*)d_out;

    // Workspace: Sc = H*NC*64*64 floats (8 MB), zc = H*NC*64 floats (128 KB).
    float* Sc = (float*)d_ws;
    float* zc = Sc + (size_t)HEADS * NCHUNK * DHEAD * DHEAD;

    dim3 grid(NCHUNK, HEADS);
    linattn_k1_chunk_stats<<<grid, 128, 0, stream>>>(K, V, Sc, zc);
    linattn_k2_scan<<<HEADS, 256, 0, stream>>>(Sc, zc);
    linattn_k3_output<<<grid, 128, 0, stream>>>(Q, K, V, Sc, zc, Out);
}
